// Pointer_58909771432744
// MI455X (gfx1250) — matrix operand order repaired and checked by reading, not compiled, not run on hardware
//
#include <hip/hip_runtime.h>
#include <hip/hip_bf16.h>
#include <stdint.h>

typedef __attribute__((ext_vector_type(16))) _Float16 v16h;
typedef __attribute__((ext_vector_type(8)))  _Float16 v8h;
typedef __attribute__((ext_vector_type(8)))  float    v8f;
typedef __attribute__((ext_vector_type(4)))  unsigned int v4u;
typedef __attribute__((ext_vector_type(8)))  int v8i;
typedef __attribute__((ext_vector_type(4)))  int v4i;

#define DEVI __device__ __forceinline__

constexpr int kB = 64;         // batch
constexpr int kS = 512;        // seq len (encoder positions and decode steps)
constexpr int kH = 512;        // hidden
constexpr int kG = 4 * kH;     // 2048 gate width
constexpr int kKcat = 2 * kH;  // 1024 concatenated [x|h] K dim

// ---------------- workspace layout (bytes) ----------------
constexpr size_t SZ_EH   = (size_t)kB * kS * kH * 2;   // encoder f16         32MB
constexpr size_t SZ_K1H  = (size_t)kB * kS * kH * 2;   // K1 f16              32MB
constexpr size_t SZ_WG   = (size_t)kG * kKcat * 2;     // [W_ih|W_hh] f16      4MB
constexpr size_t SZ_WH   = (size_t)kH * kH * 2;        // W1h/W2h/Winh       512KB
constexpr size_t SZ_BG   = (size_t)kG * 4;
constexpr size_t SZ_Q    = (size_t)kS * kB * kH * 4;   // all q_t f32         64MB
constexpr size_t SZ_XH   = (size_t)kB * kKcat * 2;
constexpr size_t SZ_C    = (size_t)kB * kH * 4;
constexpr size_t SZ_G    = (size_t)kB * kG * 4;

constexpr size_t OFF_EH   = 0;
constexpr size_t OFF_K1H  = OFF_EH   + SZ_EH;
constexpr size_t OFF_WG   = OFF_K1H  + SZ_K1H;
constexpr size_t OFF_W1H  = OFF_WG   + SZ_WG;
constexpr size_t OFF_W2H  = OFF_W1H  + SZ_WH;
constexpr size_t OFF_WINH = OFF_W2H  + SZ_WH;
constexpr size_t OFF_BG   = OFF_WINH + SZ_WH;
constexpr size_t OFF_Q    = OFF_BG   + SZ_BG;
constexpr size_t OFF_XH0  = OFF_Q    + SZ_Q;
constexpr size_t OFF_XH1  = OFF_XH0  + SZ_XH;
constexpr size_t OFF_C    = OFF_XH1  + SZ_XH;
constexpr size_t OFF_G    = OFF_C    + SZ_C;
constexpr size_t OFF_SYNC = OFF_G    + SZ_G;

// ---------------- math helpers ----------------
DEVI float sigmoidf_fast(float x) {
  return __builtin_amdgcn_rcpf(1.0f + __builtin_amdgcn_exp2f(-1.44269504f * x));
}
DEVI float fast_tanhf(float x) {
#if __has_builtin(__builtin_amdgcn_tanhf)
  return __builtin_amdgcn_tanhf(x);
#elif __has_builtin(__builtin_amdgcn_tanh_f32)
  return __builtin_amdgcn_tanh_f32(x);
#else
  float r; asm("v_tanh_f32 %0, %1" : "=v"(r) : "v"(x)); return r;
#endif
}

// ---------------- WMMA fragment helpers (ISA 7.12.2 layouts, wave32) ----------------
// A (16x32 f16): lane L row M=L%16; lanes 0-15: K {kb..kb+7, kb+16..kb+23};
//                lanes 16-31: K {kb+8..kb+15, kb+24..kb+31}.
DEVI v16h load_frag_a(const _Float16* base, int rstride, int tm, int kb, int lane) {
  int m  = tm + (lane & 15);
  int hi = (lane >> 4) & 1;
  const _Float16* p = base + (size_t)m * rstride + kb + hi * 8;
  v8h x0 = *(const v8h*)p;
  v8h x1 = *(const v8h*)(p + 16);
  return __builtin_shufflevector(x0, x1, 0,1,2,3,4,5,6,7,8,9,10,11,12,13,14,15);
}
// B (32x16 f16): lane L column N=L%16; lanes 0-15: K {kb..kb+7, kb+16..kb+23}; lanes 16-31: K {kb+8..kb+15, kb+24..kb+31}.
// Every GEMM here is x @ W^T, so B-column n == row n of the row-major weight matrix.
DEVI v16h load_frag_b(const _Float16* wbase, int rstride, int tn, int kb, int lane) {
  int n  = tn + (lane & 15);
  int hi = (lane >> 4) & 1;
  const _Float16* p = wbase + (size_t)n * rstride + kb + hi * 8;
  v8h x0 = *(const v8h*)p;
  v8h x1 = *(const v8h*)(p + 16);
  return __builtin_shufflevector(x0, x1, 0,1,2,3,4,5,6,7,8,9,10,11,12,13,14,15);
}
DEVI v8f wmma_f16(v16h a, v16h b, v8f c) {
  return __builtin_amdgcn_wmma_f32_16x16x32_f16(false, a, false, b, (short)0, c, false, false);
}
// C/D (16x16 f32): lane L column N=L%16, VGPR r row M = r + 8*(L/16).
DEVI void store_tile_f32(float* out, int rstride, int tm, int tn, int lane, v8f c) {
  int n  = tn + (lane & 15);
  int mb = tm + ((lane >> 4) & 1) * 8;
#pragma unroll
  for (int r = 0; r < 8; ++r) out[(size_t)(mb + r) * rstride + n] = c[r];
}

// ---------------- device-wide barrier (persistent kernel) ----------------
DEVI void gbar(unsigned* cnt, unsigned* gen) {
  __threadfence();
  __syncthreads();
  if (threadIdx.x == 0) {
    unsigned g = __hip_atomic_load(gen, __ATOMIC_RELAXED, __HIP_MEMORY_SCOPE_AGENT);
    unsigned a = __hip_atomic_fetch_add(cnt, 1u, __ATOMIC_ACQ_REL, __HIP_MEMORY_SCOPE_AGENT);
    if (a == gridDim.x - 1) {
      __hip_atomic_store(cnt, 0u, __ATOMIC_RELAXED, __HIP_MEMORY_SCOPE_AGENT);
      __hip_atomic_fetch_add(gen, 1u, __ATOMIC_RELEASE, __HIP_MEMORY_SCOPE_AGENT);
    } else {
      while (__hip_atomic_load(gen, __ATOMIC_ACQUIRE, __HIP_MEMORY_SCOPE_AGENT) == g)
        __builtin_amdgcn_s_sleep(1);
    }
  }
  __syncthreads();
  __threadfence();
}

// ---------------- setup kernels ----------------
__global__ void k_cvt(const float* __restrict__ s, _Float16* __restrict__ d, int n) {
  int st = gridDim.x * blockDim.x;
  for (int i = blockIdx.x * blockDim.x + threadIdx.x; i < n; i += st)
    d[i] = (_Float16)s[i];
}

__global__ void k_build_wg(const float* __restrict__ Wih, const float* __restrict__ Whh,
                           _Float16* __restrict__ Wg) {
  int n = kG * kKcat;
  int st = gridDim.x * blockDim.x;
  for (int i = blockIdx.x * blockDim.x + threadIdx.x; i < n; i += st) {
    int r = i >> 10, k = i & 1023;
    float v = (k < kH) ? Wih[(size_t)r * kH + k] : Whh[(size_t)r * kH + (k - kH)];
    Wg[i] = (_Float16)v;
  }
}

__global__ void k_init(const float* __restrict__ b_ih, const float* __restrict__ b_hh,
                       const float* __restrict__ b_in, const float* __restrict__ h0,
                       const float* __restrict__ c0, float* __restrict__ bg,
                       _Float16* __restrict__ xh0, float* __restrict__ Cb,
                       unsigned* __restrict__ sync) {
  int gtid = blockIdx.x * blockDim.x + threadIdx.x;
  int st = gridDim.x * blockDim.x;
  for (int i = gtid; i < kG; i += st) bg[i] = b_ih[i] + b_hh[i];
  for (int i = gtid; i < kB * kKcat; i += st) {
    int m = i >> 10, k = i & 1023;
    // x'_0 = sigmoid(0 @ W_in^T + b_in) = sigmoid(b_in);  h part = h0
    float v = (k < kH) ? sigmoidf_fast(b_in[k]) : h0[(size_t)m * kH + (k - kH)];
    xh0[i] = (_Float16)v;
  }
  for (int i = gtid; i < kB * kH; i += st) Cb[i] = c0[i];
  if (gtid == 0) { sync[0] = 0u; sync[1] = 0u; }
}

// ---------------- kernel 1: K1 = enc @ W1^T + b1 (f16 out), pipelined ----------------
__global__ void __launch_bounds__(256) k_k1(const _Float16* __restrict__ Eh,
                                            const _Float16* __restrict__ W1h,
                                            const float* __restrict__ b1,
                                            _Float16* __restrict__ K1h) {
  int w    = (blockIdx.x * 256 + threadIdx.x) >> 5;  // strip id, 16384 total
  int lane = threadIdx.x & 31;
  int mi = w >> 3;            // 0..2047 (16-row tile of B*S)
  int nb = (w & 7) * 64;      // 64-col strip of H
  v8f acc0 = {}, acc1 = {}, acc2 = {}, acc3 = {};
  v16h a_c  = load_frag_a(Eh, kH, mi * 16, 0, lane);
  v16h b_c0 = load_frag_b(W1h, kH, nb +  0, 0, lane);
  v16h b_c1 = load_frag_b(W1h, kH, nb + 16, 0, lane);
  v16h b_c2 = load_frag_b(W1h, kH, nb + 32, 0, lane);
  v16h b_c3 = load_frag_b(W1h, kH, nb + 48, 0, lane);
  for (int kb = 0; kb < kH; kb += 32) {
    v16h a_n = a_c, b_n0 = b_c0, b_n1 = b_c1, b_n2 = b_c2, b_n3 = b_c3;
    int kn = kb + 32;
    if (kn < kH) {  // prefetch next k-slice before consuming current one
      a_n  = load_frag_a(Eh, kH, mi * 16, kn, lane);
      b_n0 = load_frag_b(W1h, kH, nb +  0, kn, lane);
      b_n1 = load_frag_b(W1h, kH, nb + 16, kn, lane);
      b_n2 = load_frag_b(W1h, kH, nb + 32, kn, lane);
      b_n3 = load_frag_b(W1h, kH, nb + 48, kn, lane);
    }
    acc0 = wmma_f16(a_c, b_c0, acc0);
    acc1 = wmma_f16(a_c, b_c1, acc1);
    acc2 = wmma_f16(a_c, b_c2, acc2);
    acc3 = wmma_f16(a_c, b_c3, acc3);
    a_c = a_n; b_c0 = b_n0; b_c1 = b_n1; b_c2 = b_n2; b_c3 = b_n3;
  }
  int mb = mi * 16 + ((lane >> 4) & 1) * 8;
  v8f* accs[4] = {&acc0, &acc1, &acc2, &acc3};
#pragma unroll
  for (int j = 0; j < 4; ++j) {
    int n = nb + j * 16 + (lane & 15);
    float bias = b1[n];
#pragma unroll
    for (int r = 0; r < 8; ++r)
      K1h[(size_t)(mb + r) * kH + n] = (_Float16)((*accs[j])[r] + bias);
  }
}

// Pipelined single-tile GEMM body used by the recurrent kernel.
DEVI v8f gemm_tile(const _Float16* abase, int astride, int tm,
                   const _Float16* wbase, int wstride, int tn,
                   int Kdim, int lane) {
  v8f acc = {};
  v16h a_c = load_frag_a(abase, astride, tm, 0, lane);
  v16h b_c = load_frag_b(wbase, wstride, tn, 0, lane);
  for (int kb = 0; kb < Kdim; kb += 32) {
    v16h a_n = a_c, b_n = b_c;
    if (kb + 32 < Kdim) {
      a_n = load_frag_a(abase, astride, tm, kb + 32, lane);
      b_n = load_frag_b(wbase, wstride, tn, kb + 32, lane);
    }
    acc = wmma_f16(a_c, b_c, acc);
    a_c = a_n; b_c = b_n;
  }
  return acc;
}

// ---------------- kernel 2: persistent recurrent LSTM chain ----------------
// Phase A(t): gates_t = [x_t|h_{t-1}] @ [W_ih;W_hh]^T + (b_ih+b_hh);  q_{t-1} = h_{t-1}@W2^T + b2
// Phase B(t): c_t,h_t elementwise; x_{t+1} = sigmoid(x_t @ W_in^T + b_in)
__global__ void __launch_bounds__(256) k_recurrent(
    const float* __restrict__ bg, const float* __restrict__ b2, const float* __restrict__ b_in,
    const _Float16* __restrict__ Wg, const _Float16* __restrict__ W2h, const _Float16* __restrict__ Winh,
    _Float16* xh0, _Float16* xh1, float* __restrict__ Cb, float* __restrict__ Gb,
    float* __restrict__ Q, unsigned* sync) {
  const int tid  = threadIdx.x;
  const int gtid = blockIdx.x * 256 + tid;
  const int w    = gtid >> 5;     // 0..511 waves
  const int lane = tid & 31;
  unsigned* cnt = sync;
  unsigned* gen = sync + 1;
  _Float16* cur = xh0;
  _Float16* nxt = xh1;

  for (int t = 0; t <= kS; ++t) {
    // ---- Phase A ----
    if (t < kS) {  // gate pre-activations: 4 x 128 tiles, K = 1024
      int mi = w >> 7, ni = w & 127;
      v8f acc = gemm_tile(cur, kKcat, mi * 16, Wg, kKcat, ni * 16, kKcat, lane);
      float bias = bg[ni * 16 + (lane & 15)];
#pragma unroll
      for (int r = 0; r < 8; ++r) acc[r] += bias;
      store_tile_f32(Gb, kG, mi * 16, ni * 16, lane, acc);
    }
    if (t >= 1 && w < 128) {  // q_{t-1} = h_{t-1} @ W2^T + b2 : 4 x 32 tiles, K = 512
      int mi = w >> 5, ni = w & 31;
      v8f acc = gemm_tile(cur + kH, kKcat, mi * 16, W2h, kH, ni * 16, kH, lane);
      float bias = b2[ni * 16 + (lane & 15)];
#pragma unroll
      for (int r = 0; r < 8; ++r) acc[r] += bias;
      store_tile_f32(Q + (size_t)(t - 1) * kB * kH, kH, mi * 16, ni * 16, lane, acc);
    }
    gbar(cnt, gen);

    // ---- Phase B ----
    if (t < kS) {
      if (w < 128) {  // x_{t+1} = sigmoid(x_t @ W_in^T + b_in)
        int mi = w >> 5, ni = w & 31;
        v8f acc = gemm_tile(cur, kKcat, mi * 16, Winh, kH, ni * 16, kH, lane);
        int n  = ni * 16 + (lane & 15);
        int mb = mi * 16 + ((lane >> 4) & 1) * 8;
        float bias = b_in[n];
#pragma unroll
        for (int r = 0; r < 8; ++r)
          nxt[(size_t)(mb + r) * kKcat + n] = (_Float16)sigmoidf_fast(acc[r] + bias);
      }
      // LSTM cell elementwise: gate order i, f, g, o
      for (int e = gtid; e < kB * kH; e += 64 * 256) {
        int m = e >> 9, n2 = e & 511;
        const float* gp = Gb + (size_t)m * kG + n2;
        float gi = gp[0], gf = gp[512], gg = gp[1024], go = gp[1536];
        float c = sigmoidf_fast(gf) * Cb[e] + sigmoidf_fast(gi) * fast_tanhf(gg);
        Cb[e] = c;
        nxt[(size_t)m * kKcat + kH + n2] = (_Float16)(sigmoidf_fast(go) * fast_tanhf(c));
      }
      gbar(cnt, gen);
    }
    _Float16* tmp = cur; cur = nxt; nxt = tmp;
  }
}

// ---------------- kernel 3: attention logits ----------------
// logits[b, t, s] = tanh(K1[b,s,:] + q_t[b,:]) . v + b_v  (masked)
// 32 s-rows of K1 staged into LDS with a 16B/row pad (bank-conflict-free interleaved
// f16 access). Staging uses the Tensor Data Mover when available: one 2D descriptor
// (data_size=4B, 256-dword rows, pad_interval=256 dwords, pad_amount=4 dwords).
constexpr int ATS  = 32;            // s-rows per block
constexpr int APAD = 8;             // f16 row padding (16 bytes)
constexpr int ASTR = kH + APAD;     // 520

#if __has_builtin(__builtin_amdgcn_tensor_load_to_lds)
#define HAVE_TDM 1
DEVI void tdm_load_k1_tile(uint32_t lds_off, const void* gaddr) {
  uint64_t ga = (uint64_t)(uintptr_t)gaddr;
  v4u g0;
  g0.x = 0x1u;                                                 // count=1, user desc
  g0.y = lds_off;                                              // lds_addr
  g0.z = (uint32_t)ga;                                         // global_addr[31:0]
  g0.w = (uint32_t)((ga >> 32) & 0x01FFFFFFu) | 0x80000000u;   // global_addr[56:32] | type=2
  v8i g1;
  g1[0] = (int)((2u << 16) | (1u << 20) | (7u << 22) | (3u << 25)); // 4B elems, pad 4dw per 256dw
  g1[1] = (int)(256u << 16);   // tensor_dim0 = 256 dwords (one K1 row)
  g1[2] = (int)(32u << 16);    // tensor_dim1 = 32 rows
  g1[3] = (int)(256u << 16);   // tile_dim0 = 256 dwords
  g1[4] = 32;                  // tile_dim1 = 32, tile_dim2 = 0
  g1[5] = 256;                 // tensor_dim0_stride = 256 dwords
  g1[6] = 0;
  g1[7] = 0;
  v4i z4 = {0, 0, 0, 0};
  v8i z8 = {0, 0, 0, 0, 0, 0, 0, 0};
  __builtin_amdgcn_tensor_load_to_lds(g0, g1, z4, z4, z8, 0);
}
#endif

__global__ void __launch_bounds__(256) k_attn(const _Float16* __restrict__ K1h,
                                              const float* __restrict__ Q,
                                              const float* __restrict__ vvec,
                                              const unsigned char* __restrict__ mask,
                                              const float* __restrict__ bvp,
                                              float* __restrict__ out) {
  __shared__ _Float16 sK[ATS * ASTR];
  __shared__ float sV[kH];
  __shared__ float sQ[kH];
  const int tid = threadIdx.x;
  const int b   = blockIdx.x >> 4;
  const int s0  = (blockIdx.x & 15) * ATS;
  const _Float16* gsrc = K1h + (size_t)(b * kS + s0) * kH;

#ifdef HAVE_TDM
  if (tid < 32) {
    tdm_load_k1_tile((uint32_t)(uintptr_t)&sK[0], gsrc);
    __builtin_amdgcn_s_wait_tensorcnt(0);
  }
#else
  for (int i = tid; i < ATS * (kH / 2); i += 256) {
    int s = i >> 8, c = i & 255;
    ((uint32_t*)(sK + s * ASTR))[c] = ((const uint32_t*)(gsrc + (size_t)s * kH))[c];
  }
#endif
  for (int i = tid; i < kH; i += 256) sV[i] = vvec[i];
  __syncthreads();

  const int sl = tid >> 3;      // 0..31: local s
  const int hq = tid & 7;       // 0..7 : h slice (interleaved h = hq + 8*i)
  const int s  = s0 + sl;
  const bool mok = mask[b * kS + s] != 0;
  const float bv = bvp[0];
  float* orow = out + (size_t)b * kS * kS + s;

  for (int t = 0; t < kS; ++t) {
    const float* qr = Q + ((size_t)t * kB + b) * kH;
    for (int i = tid; i < kH; i += 256) sQ[i] = qr[i];
    if (t + 1 < kS) __builtin_prefetch(Q + ((size_t)(t + 1) * kB + b) * kH, 0, 1);
    __syncthreads();
    const _Float16* kp = sK + sl * ASTR;
    float acc = 0.f;
#pragma unroll 8
    for (int i = 0; i < kH / 8; ++i) {
      int h = hq + 8 * i;
      acc += fast_tanhf((float)kp[h] + sQ[h]) * sV[h];
    }
    acc += __shfl_xor(acc, 1, 32);
    acc += __shfl_xor(acc, 2, 32);
    acc += __shfl_xor(acc, 4, 32);
    if (hq == 0) {
      float u = acc + bv;
      if (!mok) u = -1e9f;
      orow[(size_t)t * kS] = u;
    }
    __syncthreads();
  }
}

// ---------------- kernel 4: argmax over s ----------------
__global__ void k_argmax(const float* __restrict__ logits, int* __restrict__ preds) {
  int r = blockIdx.x * 256 + threadIdx.x;
  if (r >= kB * kS) return;
  const float* row = logits + (size_t)r * kS;
  float best = row[0]; int bi = 0;
  for (int s2 = 1; s2 < kS; ++s2) {
    float v = row[s2];
    if (v > best) { best = v; bi = s2; }
  }
  preds[r] = bi;
}

// ---------------- launcher ----------------
extern "C" void kernel_launch(void* const* d_in, const int* in_sizes, int n_in,
                              void* d_out, int out_size, void* d_ws, size_t ws_size,
                              hipStream_t stream) {
  (void)in_sizes; (void)n_in; (void)out_size; (void)ws_size;
  const float* enc  = (const float*)d_in[0];
  const unsigned char* mask = (const unsigned char*)d_in[1];
  const float* h0   = (const float*)d_in[2];
  const float* c0   = (const float*)d_in[3];
  const float* W_in = (const float*)d_in[4];
  const float* b_in = (const float*)d_in[5];
  const float* W_ih = (const float*)d_in[6];
  const float* b_ih = (const float*)d_in[7];
  const float* W_hh = (const float*)d_in[8];
  const float* b_hh = (const float*)d_in[9];
  const float* W1   = (const float*)d_in[10];
  const float* b1   = (const float*)d_in[11];
  const float* W2   = (const float*)d_in[12];
  const float* b2   = (const float*)d_in[13];
  const float* vv   = (const float*)d_in[14];
  const float* bv   = (const float*)d_in[15];

  char* wsb = (char*)d_ws;
  _Float16* Eh   = (_Float16*)(wsb + OFF_EH);
  _Float16* K1h  = (_Float16*)(wsb + OFF_K1H);
  _Float16* Wg   = (_Float16*)(wsb + OFF_WG);
  _Float16* W1h  = (_Float16*)(wsb + OFF_W1H);
  _Float16* W2h  = (_Float16*)(wsb + OFF_W2H);
  _Float16* Winh = (_Float16*)(wsb + OFF_WINH);
  float*    bg   = (float*)(wsb + OFF_BG);
  float*    Qb   = (float*)(wsb + OFF_Q);
  _Float16* xh0  = (_Float16*)(wsb + OFF_XH0);
  _Float16* xh1  = (_Float16*)(wsb + OFF_XH1);
  float*    Cb   = (float*)(wsb + OFF_C);
  float*    Gb   = (float*)(wsb + OFF_G);
  unsigned* sync = (unsigned*)(wsb + OFF_SYNC);

  float* logits = (float*)d_out;
  int*   preds  = (int*)d_out + (size_t)kB * kS * kS;

  k_cvt<<<2048, 256, 0, stream>>>(enc, Eh, kB * kS * kH);
  k_cvt<<<256, 256, 0, stream>>>(W1, W1h, kH * kH);
  k_cvt<<<256, 256, 0, stream>>>(W2, W2h, kH * kH);
  k_cvt<<<256, 256, 0, stream>>>(W_in, Winh, kH * kH);
  k_build_wg<<<2048, 256, 0, stream>>>(W_ih, W_hh, Wg);
  k_init<<<256, 256, 0, stream>>>(b_ih, b_hh, b_in, h0, c0, bg, xh0, Cb, sync);

  k_k1<<<2048, 256, 0, stream>>>(Eh, W1h, b1, K1h);                 // 16384 wave-strips
  k_recurrent<<<64, 256, 0, stream>>>(bg, b2, b_in, Wg, W2h, Winh,
                                      xh0, xh1, Cb, Gb, Qb, sync);  // persistent
  k_attn<<<kB * (kS / ATS), 256, 0, stream>>>(K1h, Qb, vv, mask, bv, logits);
  k_argmax<<<(kB * kS + 255) / 256, 256, 0, stream>>>(logits, preds);
}
